// Conv1dRNN_455266533714
// MI455X (gfx1250) — matrix operand order repaired and checked by reading, not compiled, not run on hardware
//
#include <hip/hip_runtime.h>

typedef __attribute__((ext_vector_type(16))) _Float16 v16h;
typedef __attribute__((ext_vector_type(8)))  _Float16 v8h;
typedef __attribute__((ext_vector_type(8)))  float    v8f;

// Problem constants: B=512, I=64, H=128, T=512, O=256, C=10, conv center tap c=1

// Branch-free tanh: 1 - 2/(1+e^{2x}); exact saturation at +/-inf, no EXEC games.
__device__ __forceinline__ float fast_tanh(float x) {
    float e = __builtin_amdgcn_exp2f(x * 2.88539008177793f);   // e^{2x}
    return 1.0f - 2.0f * __builtin_amdgcn_rcpf(e + 1.0f);
}
__device__ __forceinline__ float fast_sigmoid(float x) {
    return __builtin_amdgcn_rcpf(1.0f + __builtin_amdgcn_exp2f(-1.44269504088896f * x));
}

// A-fragment loader for V_WMMA_F32_16X16X32_F16, per CDNA5 ISA 16-bit A 16x32 layout:
// lane = 16*g + r holds row M=r; halves 0..7 = K[g*8 .. g*8+7], halves 8..15 = K[16+g*8 ..].
__device__ __forceinline__ v16h afrag(const _Float16* buf, int pitch, int r, int g, int kbase) {
    const _Float16* p = buf + r * pitch + kbase + g * 8;
    v8h lo = *(const v8h*)p;
    v8h hi = *(const v8h*)(p + 16);
    v16h o;
#pragma unroll
    for (int j = 0; j < 8; ++j) { o[j] = lo[j]; o[8 + j] = hi[j]; }
    return o;
}

__device__ __forceinline__ v16h bfrag(const _Float16* buf, int pitch, int n, int g, int kbase) {
    v16h o;
#pragma unroll
    for (int j = 0; j < 8; ++j) {
        o[j]     = buf[(kbase + g * 8 + j) * pitch + n];
        o[8 + j] = buf[(kbase + 16 + g * 8 + j) * pitch + n];
    }
    return o;
}

__global__ __launch_bounds__(256) void rnn_scan_kernel(
    const float* __restrict__ x,
    const float* __restrict__ Wih0, const float* __restrict__ bih0,
    const float* __restrict__ Whh0, const float* __restrict__ bhh0,
    const float* __restrict__ Wih1, const float* __restrict__ bih1,
    const float* __restrict__ Whh1, const float* __restrict__ bhh1,
    const float* __restrict__ Wc,   const float* __restrict__ bc,
    const float* __restrict__ Wl,   const float* __restrict__ bl,
    float* __restrict__ out)
{
    // LDS layout (units: _Float16)
    __shared__ __attribute__((aligned(32))) _Float16 s[67584];
    __shared__ float zb[16 * 256];
    _Float16* s_wi0 = s;               //  64x128  [0, 8192)
    _Float16* s_wh0 = s + 8192;        // 128x128  [8192, 24576)
    _Float16* s_wi1 = s + 24576;       // 128x128
    _Float16* s_wh1 = s + 40960;       // 128x128  [.., 57344)
    _Float16* s_x   = s + 57344;       // 2 x 16x64
    _Float16* s_h0  = s + 59392;       // 2 x 16x128
    _Float16* s_h1  = s + 63488;       // 2 x 16x128
    _Float16* s_wc  = s + 8192;        // 128x256, reuses wh0+wi1 region after scan

    const int tid  = threadIdx.x;
    const int lane = tid & 31;
    const int wave = tid >> 5;         // 0..7 : owns output channels [wave*16, wave*16+16)
    const int r    = lane & 15;        // A row / C-D N-lane
    const int g    = lane >> 4;
    const int b0   = blockIdx.x * 16;  // batch tile base
    const int n0w  = wave * 16;
    const int nl   = lane & 15;

    // ---- one-time: stage center-tap weights into LDS as f16, [K][N] row-major ----
    for (int idx = tid; idx < 64 * 128; idx += 256) {
        int i = idx >> 7, h = idx & 127;
        s_wi0[idx] = (_Float16)Wih0[h * 192 + i * 3 + 1];
    }
    for (int idx = tid; idx < 128 * 128; idx += 256) {
        int k = idx >> 7, h = idx & 127;
        s_wh0[idx] = (_Float16)Whh0[h * 384 + k * 3 + 1];
        s_wi1[idx] = (_Float16)Wih1[h * 384 + k * 3 + 1];
        s_wh1[idx] = (_Float16)Whh1[h * 384 + k * 3 + 1];
    }
    for (int idx = tid; idx < 2 * 2048; idx += 256) {
        s_h0[idx] = (_Float16)0.f;
        s_h1[idx] = (_Float16)0.f;
    }
    // x tile for t=0 into x buffer 0: thread owns 4 consecutive i of one row
    const int pr = (tid * 4) >> 6;          // row 0..15
    const int pi = (tid * 4) & 63;          // i base
    const float* xrowp = x + (size_t)(b0 + pr) * 32768 + (size_t)pi * 512;
#pragma unroll
    for (int e = 0; e < 4; ++e)
        s_x[pr * 64 + pi + e] = (_Float16)xrowp[e * 512 + 0];

    const float bias0 = bih0[n0w + nl] + bhh0[n0w + nl];
    const float bias1 = bih1[n0w + nl] + bhh1[n0w + nl];
    __syncthreads();

    // ---- weight B-fragments: stationary in registers for the whole scan ----
    // B layout: lane L holds column N=L%16, K in the same order as the A fragment.
    v16h fwi0[2], fwh0[4], fwi1[4], fwh1[4];
#pragma unroll
    for (int kt = 0; kt < 2; ++kt)
        fwi0[kt] = bfrag(s_wi0, 128, n0w + nl, g, kt * 32);
#pragma unroll
    for (int kt = 0; kt < 4; ++kt) {
        fwh0[kt] = bfrag(s_wh0, 128, n0w + nl, g, kt * 32);
        fwi1[kt] = bfrag(s_wi1, 128, n0w + nl, g, kt * 32);
        fwh1[kt] = bfrag(s_wh1, 128, n0w + nl, g, kt * 32);
    }

    // ---- sequential scan: 14 WMMAs / wave / step, 2 barriers / step ----
    for (int t = 0; t < 512; ++t) {
        const int cb = t & 1, nb = cb ^ 1;

        // prefetch x(t+1) into registers (off the critical WMMA chain)
        float px[4];
        if (t < 511) {
#pragma unroll
            for (int e = 0; e < 4; ++e) px[e] = xrowp[e * 512 + t + 1];
        }

        // phase 1: nh0 = tanh(x_t @ wi0^T + h0 @ wh0^T + b)
        // preload all A-fragments first so ds_load_b128s overlap the WMMA chain
        v16h a1[6];
#pragma unroll
        for (int kt = 0; kt < 2; ++kt) a1[kt]     = afrag(s_x  + cb * 1024,  64, r, g, kt * 32);
#pragma unroll
        for (int kt = 0; kt < 4; ++kt) a1[2 + kt] = afrag(s_h0 + cb * 2048, 128, r, g, kt * 32);

        v8f acc = {};
        acc = __builtin_amdgcn_wmma_f32_16x16x32_f16(false, a1[0], false, fwi0[0], (short)0, acc, false, false);
        acc = __builtin_amdgcn_wmma_f32_16x16x32_f16(false, a1[1], false, fwi0[1], (short)0, acc, false, false);
#pragma unroll
        for (int kt = 0; kt < 4; ++kt)
            acc = __builtin_amdgcn_wmma_f32_16x16x32_f16(false, a1[2 + kt], false, fwh0[kt], (short)0, acc, false, false);

#pragma unroll
        for (int e = 0; e < 8; ++e) {
            float v = fast_tanh(acc[e] + bias0);
            s_h0[nb * 2048 + (e + 8 * g) * 128 + n0w + nl] = (_Float16)v;  // M=e+8g, N=n0w+nl
        }
        __syncthreads();

        // phase 2: nh1 = tanh(nh0 @ wi1^T + h1 @ wh1^T + b)
        v16h a2[8];
#pragma unroll
        for (int kt = 0; kt < 4; ++kt) {
            a2[kt]     = afrag(s_h0 + nb * 2048, 128, r, g, kt * 32);
            a2[4 + kt] = afrag(s_h1 + cb * 2048, 128, r, g, kt * 32);
        }
        v8f acc1 = {};
#pragma unroll
        for (int kt = 0; kt < 4; ++kt)
            acc1 = __builtin_amdgcn_wmma_f32_16x16x32_f16(false, a2[kt], false, fwi1[kt], (short)0, acc1, false, false);
#pragma unroll
        for (int kt = 0; kt < 4; ++kt)
            acc1 = __builtin_amdgcn_wmma_f32_16x16x32_f16(false, a2[4 + kt], false, fwh1[kt], (short)0, acc1, false, false);

#pragma unroll
        for (int e = 0; e < 8; ++e) {
            float v = fast_tanh(acc1[e] + bias1);
            s_h1[nb * 2048 + (e + 8 * g) * 128 + n0w + nl] = (_Float16)v;
        }
        // commit x(t+1) prefetch to the other x buffer
        if (t < 511) {
#pragma unroll
            for (int e = 0; e < 4; ++e)
                s_x[nb * 1024 + pr * 64 + pi + e] = (_Float16)px[e];
        }
        __syncthreads();
    }
    // final h1 lives in buffer (512 & 1) == 0, i.e. s_h1[0..]

    // ---- classifier: z = silu(h1 @ wc^T + bc) ----
    for (int idx = tid; idx < 128 * 256; idx += 256) {       // stage wc [K=128][N=256]
        int k = idx >> 8, o = idx & 255;
        s_wc[idx] = (_Float16)Wc[o * 384 + k * 3 + 1];
    }
    __syncthreads();

#pragma unroll
    for (int sub = 0; sub < 2; ++sub) {
        const int n0 = (wave * 2 + sub) * 16;
        v8f acc = {};
#pragma unroll
        for (int kt = 0; kt < 4; ++kt) {
            v16h a = afrag(s_h1, 128, r, g, kt * 32);
            v16h b = bfrag(s_wc, 256, n0 + nl, g, kt * 32);
            acc = __builtin_amdgcn_wmma_f32_16x16x32_f16(false, a, false, b,
                                                         (short)0, acc, false, false);
        }
        const float bcv = bc[n0 + nl];
#pragma unroll
        for (int e = 0; e < 8; ++e) {
            float z = acc[e] + bcv;
            z = z * fast_sigmoid(z);                          // SiLU
            zb[(e + 8 * g) * 256 + n0 + nl] = z;
        }
    }
    __syncthreads();

    // ---- tiny head: logits = z @ Wl^T + bl  (16 x 256 @ 256 x 10) ----
    if (tid < 160) {
        const int m = tid / 10, c = tid % 10;
        float sum = bl[c];
        for (int o = 0; o < 256; ++o) sum += zb[m * 256 + o] * Wl[c * 256 + o];
        out[(b0 + m) * 10 + c] = sum;
    }
}

extern "C" void kernel_launch(void* const* d_in, const int* in_sizes, int n_in,
                              void* d_out, int out_size, void* d_ws, size_t ws_size,
                              hipStream_t stream) {
    const float* x    = (const float*)d_in[0];
    const float* Wih0 = (const float*)d_in[1];
    const float* bih0 = (const float*)d_in[2];
    const float* Whh0 = (const float*)d_in[3];
    const float* bhh0 = (const float*)d_in[4];
    const float* Wih1 = (const float*)d_in[5];
    const float* bih1 = (const float*)d_in[6];
    const float* Whh1 = (const float*)d_in[7];
    const float* bhh1 = (const float*)d_in[8];
    const float* Wc   = (const float*)d_in[9];
    const float* bc   = (const float*)d_in[10];
    const float* Wl   = (const float*)d_in[11];
    const float* bl   = (const float*)d_in[12];

    rnn_scan_kernel<<<dim3(32), dim3(256), 0, stream>>>(
        x, Wih0, bih0, Whh0, bhh0, Wih1, bih1, Whh1, bhh1, Wc, bc, Wl, bl,
        (float*)d_out);
}
